// Lstmhidden_67087389164149
// MI455X (gfx1250) — matrix operand order repaired and checked by reading, not compiled, not run on hardware
//
#include <hip/hip_runtime.h>
#include <cstddef>
#include <cstdint>

#define SLEN    128
#define BATCH   256
#define DIM     900
#define DP      928          // DIM padded to multiple of 32 (bf16 K-chunks)
#define GATES   3600
#define KCHUNKS 29           // 29 * 32 == DP
#define NTILE   80           // N columns per block (3600 = 45 * 80)
#define BSTRIDE 40           // LDS row stride in elements (16B aligned, bank-spread)

typedef __bf16 bf16_t;
typedef __bf16 v16bf __attribute__((ext_vector_type(16)));
typedef __bf16 v8bf  __attribute__((ext_vector_type(8)));
typedef float  v8f   __attribute__((ext_vector_type(8)));

static __device__ __forceinline__ v16bf cat16(v8bf lo, v8bf hi) {
    return __builtin_shufflevector(lo, hi, 0,1,2,3,4,5,6,7,8,9,10,11,12,13,14,15);
}

// ---------------- prologue kernels ----------------

__global__ void lengths_kernel(const float* __restrict__ umask, int* __restrict__ L) {
    int b = threadIdx.x;                       // 256 threads, one per batch row
    float acc = 0.f;
    for (int t = 0; t < SLEN; ++t) acc += umask[b * SLEN + t];
    L[b] = (int)(acc + 0.5f);
}

// emotions_feat [S,B,D] fp32 -> padded bf16 [S,B,DP] (cols 900..927 zero)
__global__ void convert_x_kernel(const float* __restrict__ X, bf16_t* __restrict__ Xb) {
    size_t idx = (size_t)blockIdx.x * blockDim.x + threadIdx.x;
    size_t total = (size_t)SLEN * BATCH * DP;
    if (idx >= total) return;
    int    k  = (int)(idx % DP);
    size_t sb = idx / DP;                      // t*BATCH + b
    float v = (k < DIM) ? X[sb * DIM + k] : 0.f;
    Xb[idx] = (bf16_t)v;
}

// pack [Wih1, Whh1, Wih2, Whh2] -> bf16 [4][G][DP] (K-padded)
__global__ void convert_w_kernel(const float* __restrict__ Wih1, const float* __restrict__ Whh1,
                                 const float* __restrict__ Wih2, const float* __restrict__ Whh2,
                                 bf16_t* __restrict__ Wall) {
    size_t idx = (size_t)blockIdx.x * blockDim.x + threadIdx.x;
    size_t per = (size_t)GATES * DP;
    if (idx >= 4 * per) return;
    int    w   = (int)(idx / per);
    size_t rem = idx - (size_t)w * per;
    int r = (int)(rem / DP);
    int k = (int)(rem % DP);
    const float* src = (w == 0) ? Wih1 : (w == 1) ? Whh1 : (w == 2) ? Wih2 : Whh2;
    float v = (k < DIM) ? src[(size_t)r * DIM + k] : 0.f;
    Wall[idx] = (bf16_t)v;
}

// ---------------- per-step fused GEMM: gates = [x,h] @ [Wih,Whh]^T + b ----------------
// grid: (45 N-blocks of 80 cols, 1, 2 directions); block 256 threads = 8 waves.
// Wave w computes C rows [w*32, w*32+32) x 80 cols = 10 WMMA tiles per K-chunk.
// Double-buffered LDS for the shared W tile; one barrier per K-chunk; all B
// fragments fetched before the WMMA burst so the matrix pipe runs stall-free.
__global__ __launch_bounds__(256)
void gemm_step_kernel(const bf16_t* __restrict__ Xb,    // [S,B,DP]
                      const bf16_t* __restrict__ Wall,  // [4,G,DP]
                      const bf16_t* __restrict__ Hb,    // [2,B,DP] hidden state
                      const bf16_t* __restrict__ Zrow,  // DP zeros (masked rows)
                      const int*    __restrict__ L,
                      const float*  __restrict__ bi1, const float* __restrict__ bh1,
                      const float*  __restrict__ bi2, const float* __restrict__ bh2,
                      float* __restrict__ gates, int s)
{
    __shared__ __align__(16) bf16_t lds_b[2][NTILE * BSTRIDE];   // 2 x 6.4 KB

    const int tid   = threadIdx.x;
    const int lane  = tid & 31;
    const int wave  = tid >> 5;        // 0..7
    const int nlo   = lane & 15;
    const int half  = lane >> 4;       // 0/1
    const int d     = blockIdx.z;
    const int nbase = blockIdx.x * NTILE;

    const int b0 = wave * 32 + nlo;    // batch row, A tile 0
    const int b1 = b0 + 16;            // batch row, A tile 1

    // Loop-invariant A row bases. Direction 1 gathers reversed rows; rows past
    // L[b] point at a zeroed row instead of per-chunk cndmask'ing the fragment.
    const bf16_t *pax0, *pax1;
    {
        int t0 = s, t1 = s;
        bool v0 = true, v1 = true;
        if (d) {
            int L0 = L[b0]; v0 = (s < L0); t0 = v0 ? (L0 - 1 - s) : 0;
            int L1 = L[b1]; v1 = (s < L1); t1 = v1 ? (L1 - 1 - s) : 0;
        }
        pax0 = v0 ? (Xb + ((size_t)t0 * BATCH + b0) * DP) : Zrow;
        pax1 = v1 ? (Xb + ((size_t)t1 * BATCH + b1) * DP) : Zrow;
    }
    const bf16_t* pah0 = Hb + ((size_t)d * BATCH + b0) * DP;
    const bf16_t* pah1 = Hb + ((size_t)d * BATCH + b1) * DP;
    const bf16_t* Bx   = Wall + (size_t)(2 * d + 0) * GATES * DP;
    const bf16_t* Bh   = Wall + (size_t)(2 * d + 1) * GATES * DP;

    // B staging: 80 rows x 32 K = 320 x 16B tasks; every thread one, tid<64 a second.
    const int sn  = tid >> 2;          // 0..63
    const int sko = (tid & 3) * 8;
    const int sn2 = sn + 64;           // 64..79 (tid < 64 only)

    auto bsrc = [&](int ck, int n) -> const bf16_t* {
        const int  ph = ck >= KCHUNKS;
        const int  kc = (ph ? ck - KCHUNKS : ck) * 32;
        const bf16_t* Bp = ph ? Bh : Bx;
        return Bp + (size_t)(nbase + n) * DP + kc + sko;
    };

    // stage chunk 0 into buffer 0
    *(v8bf*)&lds_b[0][sn * BSTRIDE + sko] = *(const v8bf*)bsrc(0, sn);
    if (tid < 64)
        *(v8bf*)&lds_b[0][sn2 * BSTRIDE + sko] = *(const v8bf*)bsrc(0, sn2);
    __syncthreads();

    v8f acc[10] = {};

    for (int ck = 0; ck < 2 * KCHUNKS; ++ck) {
        const int cur = ck & 1;
        const int nxt = cur ^ 1;
        const int ph  = ck >= KCHUNKS;
        const int kc  = (ph ? ck - KCHUNKS : ck) * 32;
        const bool more = (ck + 1) < 2 * KCHUNKS;

        // issue next chunk's global loads early (latency hidden by the WMMAs)
        v8bf st0 = {}, st1 = {};
        if (more) {
            st0 = *(const v8bf*)bsrc(ck + 1, sn);
            if (tid < 64) st1 = *(const v8bf*)bsrc(ck + 1, sn2);
        }

        // A fragments, ISA 16x32 bf16 layout: VGPR0-3 K=half*8+0..7, VGPR4-7 K=16+half*8+0..7
        const bf16_t* q0 = (ph ? pah0 : pax0) + kc + half * 8;
        const bf16_t* q1 = (ph ? pah1 : pax1) + kc + half * 8;
        v16bf a0 = cat16(*(const v8bf*)q0, *(const v8bf*)(q0 + 16));
        v16bf a1 = cat16(*(const v8bf*)q1, *(const v8bf*)(q1 + 16));
        __builtin_prefetch(q0 + 32, 0, 3);     // prefetch next K chunk
        __builtin_prefetch(q1 + 32, 0, 3);

        // fetch all 5 B fragments first (clause-able ds_load_b128 stream, one wait)
        v16bf bm[5];
#pragma unroll
        for (int t = 0; t < 5; ++t) {
            // B fragment 32x16: lane n = lane%16, VGPR0-3 K=half*8+0..7, VGPR4-7 K=16+half*8+0..7
            const bf16_t* qb = &lds_b[cur][(t * 16 + nlo) * BSTRIDE + half * 8];
            bm[t] = cat16(*(const v8bf*)qb, *(const v8bf*)(qb + 16));
        }

        // WMMA burst: 10 independent accumulators, no DS waits in between
#pragma unroll
        for (int t = 0; t < 5; ++t) {
            acc[2 * t]     = __builtin_amdgcn_wmma_f32_16x16x32_bf16(false, a0, false, bm[t], (short)0, acc[2 * t],     false, false);
            acc[2 * t + 1] = __builtin_amdgcn_wmma_f32_16x16x32_bf16(false, a1, false, bm[t], (short)0, acc[2 * t + 1], false, false);
        }

        // store next chunk into the other LDS buffer, then one barrier
        if (more) {
            *(v8bf*)&lds_b[nxt][sn * BSTRIDE + sko] = st0;
            if (tid < 64) *(v8bf*)&lds_b[nxt][sn2 * BSTRIDE + sko] = st1;
        }
        __syncthreads();
    }

    // Epilogue: add biases, store gates fp32 [B, G].
    const float* bip = d ? bi2 : bi1;
    const float* bhp = d ? bh2 : bh1;
    float* Gp = gates + (size_t)d * BATCH * GATES;
    const int m0 = wave * 32 + half * 8;       // C layout: VGPR r -> row half*8+r, col = lane%16
#pragma unroll
    for (int t = 0; t < 5; ++t) {
        const int n = nbase + t * 16 + nlo;
        const float bias = bip[n] + bhp[n];
#pragma unroll
        for (int r = 0; r < 8; ++r) {
            Gp[(size_t)(m0 + r)      * GATES + n] = acc[2 * t][r]     + bias;
            Gp[(size_t)(m0 + 16 + r) * GATES + n] = acc[2 * t + 1][r] + bias;
        }
    }
}

// ---------------- LSTM cell nonlinearity + cumsum output ----------------
__global__ void cell_update_kernel(const float* __restrict__ gates,
                                   float* __restrict__ Hf, float* __restrict__ Cf,
                                   float* __restrict__ Hs, bf16_t* __restrict__ Hb,
                                   float* __restrict__ out, float* __restrict__ back, int s)
{
    int idx = blockIdx.x * blockDim.x + threadIdx.x;
    if (idx >= 2 * BATCH * DIM) return;
    int d   = idx / (BATCH * DIM);
    int rem = idx - d * (BATCH * DIM);
    int b   = rem / DIM;
    int j   = rem - b * DIM;

    const float* g = gates + (size_t)d * BATCH * GATES + (size_t)b * GATES;
    float ig = g[j];
    float fg = g[DIM + j];
    float gg = g[2 * DIM + j];
    float og = g[3 * DIM + j];

    float h  = Hf[idx];
    float c  = Cf[idx];
    float hs = Hs[idx] + h;                     // cumsum of hidden states 0..s-1

    if (d == 0) out [(size_t)s * BATCH * DIM + rem] = hs;   // front_sum straight to output
    else        back[(size_t)s * BATCH * DIM + rem] = hs;   // back_sum, reversed later

    float i_s = 1.f / (1.f + __expf(-ig));
    float f_s = 1.f / (1.f + __expf(-fg));
    float o_s = 1.f / (1.f + __expf(-og));
    c = f_s * c + i_s * tanhf(gg);
    h = o_s * tanhf(c);

    Hf[idx] = h; Cf[idx] = c; Hs[idx] = hs;
    Hb[(size_t)d * BATCH * DP + (size_t)b * DP + j] = (bf16_t)h;   // padded cols stay zero
}

// ---------------- un-reverse back_sum and add into output ----------------
__global__ void back_add_kernel(const float* __restrict__ back, const int* __restrict__ L,
                                float* __restrict__ out)
{
    size_t idx = (size_t)blockIdx.x * blockDim.x + threadIdx.x;
    size_t total = (size_t)SLEN * BATCH * DIM;
    if (idx >= total) return;
    int    j  = (int)(idx % DIM);
    size_t tb = idx / DIM;
    int b = (int)(tb % BATCH);
    int t = (int)(tb / BATCH);
    int Lb = L[b];
    float v = 0.f;
    if (t < Lb) v = back[((size_t)(Lb - 1 - t) * BATCH + b) * DIM + j];
    out[idx] += v;
}

// ---------------- host side ----------------
extern "C" void kernel_launch(void* const* d_in, const int* in_sizes, int n_in,
                              void* d_out, int out_size, void* d_ws, size_t ws_size,
                              hipStream_t stream)
{
    (void)in_sizes; (void)n_in; (void)out_size; (void)ws_size;

    const float* X     = (const float*)d_in[0];
    const float* umask = (const float*)d_in[1];
    const float* Wih1  = (const float*)d_in[2];
    const float* Whh1  = (const float*)d_in[3];
    const float* bih1  = (const float*)d_in[4];
    const float* bhh1  = (const float*)d_in[5];
    const float* Wih2  = (const float*)d_in[6];
    const float* Whh2  = (const float*)d_in[7];
    const float* bih2  = (const float*)d_in[8];
    const float* bhh2  = (const float*)d_in[9];
    float* out = (float*)d_out;

    // workspace carving (~220 MB total)
    char* base = (char*)d_ws;
    size_t off = 0;
    auto carve = [&](size_t bytes) -> void* {
        void* r = base + off;
        off += (bytes + 255) & ~(size_t)255;
        return r;
    };
    bf16_t* Xb    = (bf16_t*)carve((size_t)SLEN * BATCH * DP * 2);   //  60.8 MB
    bf16_t* Wall  = (bf16_t*)carve((size_t)4 * GATES * DP * 2);      //  26.7 MB
    float*  gates = (float*) carve((size_t)2 * BATCH * GATES * 4);   //   7.4 MB
    bf16_t* Hb    = (bf16_t*)carve((size_t)2 * BATCH * DP * 2);      //   0.95 MB
    float*  Hf    = (float*) carve((size_t)2 * BATCH * DIM * 4);     //   1.8 MB
    float*  Cf    = (float*) carve((size_t)2 * BATCH * DIM * 4);
    float*  Hs    = (float*) carve((size_t)2 * BATCH * DIM * 4);
    float*  back  = (float*) carve((size_t)SLEN * BATCH * DIM * 4);  // 118 MB
    bf16_t* Zrow  = (bf16_t*)carve((size_t)DP * 2);
    int*    L     = (int*)   carve(BATCH * 4);

    // zero recurrent state (also zeroes Hb's K-padding and Zrow)
    hipMemsetAsync(Hb, 0, (size_t)2 * BATCH * DP * 2, stream);
    hipMemsetAsync(Hf, 0, (size_t)2 * BATCH * DIM * 4, stream);
    hipMemsetAsync(Cf, 0, (size_t)2 * BATCH * DIM * 4, stream);
    hipMemsetAsync(Hs, 0, (size_t)2 * BATCH * DIM * 4, stream);
    hipMemsetAsync(Zrow, 0, (size_t)DP * 2, stream);

    lengths_kernel<<<1, BATCH, 0, stream>>>(umask, L);
    {
        size_t tot = (size_t)SLEN * BATCH * DP;
        convert_x_kernel<<<(int)((tot + 255) / 256), 256, 0, stream>>>(X, Xb);
    }
    {
        size_t tot = (size_t)4 * GATES * DP;
        convert_w_kernel<<<(int)((tot + 255) / 256), 256, 0, stream>>>(Wih1, Whh1, Wih2, Whh2, Wall);
    }

    for (int s = 0; s < SLEN; ++s) {
        gemm_step_kernel<<<dim3(GATES / NTILE, 1, 2), 256, 0, stream>>>(
            Xb, Wall, Hb, Zrow, L, bih1, bhh1, bih2, bhh2, gates, s);
        cell_update_kernel<<<(2 * BATCH * DIM + 255) / 256, 256, 0, stream>>>(
            gates, Hf, Cf, Hs, Hb, out, back, s);
    }
    {
        size_t tot = (size_t)SLEN * BATCH * DIM;
        back_add_kernel<<<(int)((tot + 255) / 256), 256, 0, stream>>>(back, L, out);
    }
}
